// SpanClassifier_90099823935812
// MI455X (gfx1250) — matrix operand order repaired and checked by reading, not compiled, not run on hardware
//
#include <hip/hip_runtime.h>
#include <hip/hip_bf16.h>
#include <math.h>

// Problem constants (B, N, H, L) = (8, 1024, 2048, 16)
#define BATCH 8
#define NSEQ  1024
#define HDIM  2048
#define LCLS  16
#define NPAIR 524800LL          // NSEQ*(NSEQ+1)/2
#define NTILE 64                // NSEQ/16
#define NTPAIR 2080             // NTILE*(NTILE+1)/2

typedef __attribute__((ext_vector_type(16))) __bf16 v16bf;
typedef __attribute__((ext_vector_type(8)))  float  v8f;
typedef __attribute__((ext_vector_type(4)))  float  v4f;   // clang vector: OK for NT builtins

// ---------------------------------------------------------------------------
// Kernel 1a: K-split partials of Mcat = [Ws@W_start ; We@W_end]  (32 x 2048)
// grid (8 hp-blocks, 2 halves, 8 k-splits) x 256 threads.
// Each block stages a 16x256 W_span chunk in LDS, reads its W rows once,
// and accumulates 16 outputs per thread -> W_start/W_end read exactly once.
// ---------------------------------------------------------------------------
__global__ __launch_bounds__(256) void build_mcat_partial(const float* __restrict__ W_start,
                                                          const float* __restrict__ W_end,
                                                          const float* __restrict__ W_span,
                                                          float* __restrict__ partial) {
    const int hp    = blockIdx.x * 256 + threadIdx.x;   // 0..2047
    const int half  = blockIdx.y;                       // 0 = start, 1 = end
    const int hbase = blockIdx.z * 256;                 // k-split origin
    const float* Wg = half ? W_end : W_start;

    __shared__ float wsh[256][16];                      // [hh][o]
    for (int t = threadIdx.x; t < 4096; t += 256) {
        int o  = t >> 8;
        int hh = t & 255;
        wsh[hh][o] = W_span[(size_t)o * (2 * HDIM) + (half ? HDIM : 0) + hbase + hh];
    }
    __syncthreads();

    float acc[16];
#pragma unroll
    for (int o = 0; o < 16; ++o) acc[o] = 0.f;

#pragma unroll 2
    for (int hh = 0; hh < 256; ++hh) {
        float w = Wg[(size_t)(hbase + hh) * HDIM + hp];
#pragma unroll
        for (int o = 0; o < 16; ++o)
            acc[o] = fmaf(wsh[hh][o], w, acc[o]);
    }

#pragma unroll
    for (int o = 0; o < 16; ++o)
        partial[((size_t)blockIdx.z * 32 + half * 16 + o) * HDIM + hp] = acc[o];
}

// ---------------------------------------------------------------------------
// Kernel 1b: reduce 8 K-split partials -> bf16 Mcat (deterministic order)
// ---------------------------------------------------------------------------
__global__ __launch_bounds__(256) void reduce_mcat(const float* __restrict__ partial,
                                                   __bf16* __restrict__ Mcat) {
    int idx = blockIdx.x * 256 + threadIdx.x;           // 0..65535  (= o*2048 + hp)
    float s = 0.f;
#pragma unroll
    for (int ks = 0; ks < 8; ++ks)
        s += partial[(size_t)ks * 65536 + idx];
    Mcat[idx] = (__bf16)s;
}

// ---------------------------------------------------------------------------
// Kernel 2: folded bias[0..31]; bias[l<16] = Ws[l]·b_start + b_span[l],
//           bias[l>=16] = We[l-16]·b_end
// ---------------------------------------------------------------------------
__global__ void build_bias(const float* __restrict__ b_start,
                           const float* __restrict__ b_end,
                           const float* __restrict__ W_span,
                           const float* __restrict__ b_span,
                           float* __restrict__ bias) {
    int l = threadIdx.x;                                // 32 threads
    float acc = 0.f;
    if (l < 16) {
        const float* wrow = W_span + (size_t)l * (2 * HDIM);
        for (int h = 0; h < HDIM; ++h) acc = fmaf(wrow[h], b_start[h], acc);
        acc += b_span[l];
    } else {
        const float* wrow = W_span + (size_t)(l - 16) * (2 * HDIM) + HDIM;
        for (int h = 0; h < HDIM; ++h) acc = fmaf(wrow[h], b_end[h], acc);
    }
    bias[l] = acc;
}

// ---------------------------------------------------------------------------
// Kernel 3: proj[g, 0..31] = x[g, :] @ Mcat^T + bias   (g = b*NSEQ + n, 8192 rows)
// One block = 16 output rows; 4 waves split K (4 x 512), deterministic LDS
// reduction.  v_wmma_f32_16x16x32_bf16, wave32 layouts used by this kernel:
//   A (16x32 bf16): lane = m + 16*half; elem e -> k = e + half*16 (contiguous run)
//   B (32x16 bf16): lane = n + 16*half; elem e -> k = e + half*16 (contiguous run)
//   C (16x16 f32) : lane L -> col n=L&15; VGPR r -> row m = r + 8*(L>>4)
// x is streamed once -> nontemporal loads.
// ---------------------------------------------------------------------------
__global__ __launch_bounds__(128) void proj_gemm(const float* __restrict__ x,
                                                 const __bf16* __restrict__ Mcat,
                                                 const float* __restrict__ bias,
                                                 float* __restrict__ proj) {
    __shared__ float lds[4][2][32][8];

    const int wave = threadIdx.x >> 5;
    const int lane = threadIdx.x & 31;
    const int half = lane >> 4;
    const int mrow = lane & 15;
    const int rowTile = blockIdx.x;                       // 0..511
    const size_t g = (size_t)rowTile * 16 + mrow;

    v8f c0 = {};
    v8f c1 = {};

    const float* aBase = x + g * HDIM;
    const int k0 = wave * 512;

#pragma unroll 2
    for (int kb = k0; kb < k0 + 512; kb += 32) {
        // ---- A tile: 16 f32 from row g (streaming/NT), converted to bf16 ----
        const float* ap = aBase + kb + half * 16;
        v4f a0 = __builtin_nontemporal_load((const v4f*)(ap + 0));
        v4f a1 = __builtin_nontemporal_load((const v4f*)(ap + 4));
        v4f a2 = __builtin_nontemporal_load((const v4f*)(ap + 8));
        v4f a3 = __builtin_nontemporal_load((const v4f*)(ap + 12));
        v16bf av;
        av[0]  = (__bf16)a0.x; av[1]  = (__bf16)a0.y; av[2]  = (__bf16)a0.z; av[3]  = (__bf16)a0.w;
        av[4]  = (__bf16)a1.x; av[5]  = (__bf16)a1.y; av[6]  = (__bf16)a1.z; av[7]  = (__bf16)a1.w;
        av[8]  = (__bf16)a2.x; av[9]  = (__bf16)a2.y; av[10] = (__bf16)a2.z; av[11] = (__bf16)a2.w;
        av[12] = (__bf16)a3.x; av[13] = (__bf16)a3.y; av[14] = (__bf16)a3.z; av[15] = (__bf16)a3.w;

        // ---- B tiles: rows 0..15 (s) and 16..31 (e) of Mcat, L2-resident ----
        const __bf16* bp0 = Mcat + (size_t)mrow * HDIM        + kb + half * 16;
        const __bf16* bp1 = Mcat + (size_t)(16 + mrow) * HDIM + kb + half * 16;
        v16bf b0 = *(const v16bf*)bp0;
        v16bf b1 = *(const v16bf*)bp1;

        c0 = __builtin_amdgcn_wmma_f32_16x16x32_bf16(false, av, false, b0,
                                                     (short)0, c0, false, false);
        c1 = __builtin_amdgcn_wmma_f32_16x16x32_bf16(false, av, false, b1,
                                                     (short)0, c1, false, false);
    }

#pragma unroll
    for (int r = 0; r < 8; ++r) {
        lds[wave][0][lane][r] = c0[r];
        lds[wave][1][lane][r] = c1[r];
    }
    __syncthreads();

    // Deterministic cross-wave (K-split) reduction + bias, write 16x32 tile.
#pragma unroll
    for (int t = threadIdx.x; t < 512; t += 128) {
        int tile = t >> 8;
        int rem  = t & 255;
        int ln   = rem >> 3;
        int r    = rem & 7;
        int m    = r + 8 * (ln >> 4);
        int ncol = tile * 16 + (ln & 15);
        float v = lds[0][tile][ln][r] + lds[1][tile][ln][r]
                + lds[2][tile][ln][r] + lds[3][tile][ln][r];
        v += bias[ncol];
        proj[((size_t)rowTile * 16 + m) * 32 + ncol] = v;
    }
}

// ---------------------------------------------------------------------------
// Kernel 4 (tiled): grid (2080 triangular 16x16 (i,j)-tiles, 8 batches).
// Thread (ti,tj) handles (i,j) = (it*16+ti, jt*16+tj); active iff j>=i.
// Tile decode is uniform per block (scalar sqrt + fixup, no 64-bit division).
// out[b, p(i,j), :] = softmax( proj[b,i,0:16] + proj[b,j,16:32] ),
// p(i,j) = i*(2N-i+1)/2 + (j-i).  Streaming NT stores (write-once, 269 MB).
// ---------------------------------------------------------------------------
__global__ __launch_bounds__(256) void epilogue(const float* __restrict__ proj,
                                                float* __restrict__ out) {
    const int t = blockIdx.x;                       // 0..2079 triangular tile id
    const int b = blockIdx.y;                       // batch

    // invert tile-triangle: f(it) = it*(2*NTILE - it + 1)/2
    float disc = 64.5f * 64.5f - 2.0f * (float)t;
    int it = (int)(64.5f - sqrtf(disc));
    if (it < 0) it = 0;
    if (it > NTILE - 1) it = NTILE - 1;
    while (it * (2 * NTILE - it + 1) / 2 > t) --it;
    while ((it + 1) * (2 * NTILE - it) / 2 <= t) ++it;
    const int jt = it + (t - it * (2 * NTILE - it + 1) / 2);

    const int i = it * 16 + (threadIdx.x >> 4);
    const int j = jt * 16 + (threadIdx.x & 15);
    if (j < i) return;                              // only diagonal tiles diverge

    const int p = i * (2 * NSEQ - i + 1) / 2 + (j - i);

    const float* s = proj + ((size_t)b * NSEQ + i) * 32;        // s features
    const float* e = proj + ((size_t)b * NSEQ + j) * 32 + 16;   // e features

    v4f s0 = *(const v4f*)(s + 0),  s1 = *(const v4f*)(s + 4);
    v4f s2 = *(const v4f*)(s + 8),  s3 = *(const v4f*)(s + 12);
    v4f e0 = *(const v4f*)(e + 0),  e1 = *(const v4f*)(e + 4);
    v4f e2 = *(const v4f*)(e + 8),  e3 = *(const v4f*)(e + 12);

    float v[16];
    v[0]=s0.x+e0.x; v[1]=s0.y+e0.y; v[2]=s0.z+e0.z; v[3]=s0.w+e0.w;
    v[4]=s1.x+e1.x; v[5]=s1.y+e1.y; v[6]=s1.z+e1.z; v[7]=s1.w+e1.w;
    v[8]=s2.x+e2.x; v[9]=s2.y+e2.y; v[10]=s2.z+e2.z; v[11]=s2.w+e2.w;
    v[12]=s3.x+e3.x; v[13]=s3.y+e3.y; v[14]=s3.z+e3.z; v[15]=s3.w+e3.w;

    float mx = v[0];
#pragma unroll
    for (int l = 1; l < 16; ++l) mx = fmaxf(mx, v[l]);
    float sum = 0.f;
#pragma unroll
    for (int l = 0; l < 16; ++l) { v[l] = __expf(v[l] - mx); sum += v[l]; }
    float inv = 1.0f / sum;

    float* o = out + ((size_t)b * NPAIR + p) * 16;
    v4f o0 = { v[0]*inv,  v[1]*inv,  v[2]*inv,  v[3]*inv  };
    v4f o1 = { v[4]*inv,  v[5]*inv,  v[6]*inv,  v[7]*inv  };
    v4f o2 = { v[8]*inv,  v[9]*inv,  v[10]*inv, v[11]*inv };
    v4f o3 = { v[12]*inv, v[13]*inv, v[14]*inv, v[15]*inv };
    __builtin_nontemporal_store(o0, (v4f*)(o + 0));
    __builtin_nontemporal_store(o1, (v4f*)(o + 4));
    __builtin_nontemporal_store(o2, (v4f*)(o + 8));
    __builtin_nontemporal_store(o3, (v4f*)(o + 12));
}

// ---------------------------------------------------------------------------
extern "C" void kernel_launch(void* const* d_in, const int* in_sizes, int n_in,
                              void* d_out, int out_size, void* d_ws, size_t ws_size,
                              hipStream_t stream) {
    const float* x       = (const float*)d_in[0];
    const float* W_start = (const float*)d_in[1];
    const float* b_start = (const float*)d_in[2];
    const float* W_end   = (const float*)d_in[3];
    const float* b_end   = (const float*)d_in[4];
    const float* W_span  = (const float*)d_in[5];
    const float* b_span  = (const float*)d_in[6];
    float* out = (float*)d_out;

    // Workspace layout (all regions fully rewritten every call):
    //   [0, 131072)                  : Mcat bf16 (32 x 2048)
    //   [131072, 131200)             : bias f32[32]
    //   [132096, 1180672)            : proj f32 (8192 x 32)          (1 MB)
    //   [1180672, 3277824)           : Mcat K-split partials f32      (2 MB)
    char*   ws      = (char*)d_ws;
    __bf16* Mcat    = (__bf16*)ws;
    float*  bias    = (float*)(ws + 131072);
    float*  proj    = (float*)(ws + 132096);
    float*  partial = (float*)(ws + 1180672);

    build_mcat_partial<<<dim3(8, 2, 8), 256, 0, stream>>>(W_start, W_end, W_span, partial);
    reduce_mcat<<<256, 256, 0, stream>>>(partial, Mcat);
    build_bias<<<1, 32, 0, stream>>>(b_start, b_end, W_span, b_span, bias);
    proj_gemm<<<512, 128, 0, stream>>>(x, Mcat, bias, proj);
    epilogue<<<dim3(NTPAIR, BATCH), 256, 0, stream>>>(proj, out);
}
